// SheafGNN_17575006175438
// MI455X (gfx1250) — matrix operand order repaired and checked by reading, not compiled, not run on hardware
//
#include <hip/hip_runtime.h>
#include <hip/hip_bf16.h>

typedef __attribute__((ext_vector_type(16))) _Float16 v16h;
typedef __attribute__((ext_vector_type(8)))  _Float16 v8h;
typedef __attribute__((ext_vector_type(8)))  float    v8f;

#define HID 128
#define DD 4

// ---------------------------------------------------------------------------
// Pack a row-major f32 weight matrix W[Kdim][Ndim] into f16 WMMA B-fragments.
// Fragment (kt, nt): 512 halfs; lane L holds 16 halfs j=0..15 with
//   k = kt*32 + (L/16)*8 + (j%8) + (j/8)*16 , n = nt*16 + (L%16)
// ---------------------------------------------------------------------------
__global__ void pack_b_f16(const float* __restrict__ W, _Float16* __restrict__ frag,
                           int Kdim, int Ndim) {
    int tid = blockIdx.x * blockDim.x + threadIdx.x;
    int total = Kdim * Ndim;
    if (tid >= total) return;
    int numNT = Ndim >> 4;
    int f = tid >> 9;            // fragment index
    int r = tid & 511;
    int lane = r >> 4;
    int j = r & 15;
    int kt = f / numNT, nt = f - kt * numNT;
    int k = kt * 32 + (lane >> 4) * 8 + (j & 7) + (j >> 3) * 16;
    int n = nt * 16 + (lane & 15);
    frag[tid] = (_Float16)W[(size_t)k * Ndim + n];
}

// ---------------------------------------------------------------------------
// Node-level GEMM: Y[M][NDIM] = X[M][KDIM] @ W + bias   (one wave per tile)
// ---------------------------------------------------------------------------
template <int KDIM, int NDIM>
__global__ void node_gemm(const float* __restrict__ X, const _Float16* __restrict__ Bfrag,
                          const float* __restrict__ bias, float* __restrict__ Y, int M) {
    const int lane = threadIdx.x & 31;
    const int wave = threadIdx.x >> 5;
    const int numNT = NDIM >> 4;
    const int numMT = (M + 15) >> 4;
    const int tile = blockIdx.x * (blockDim.x >> 5) + wave;
    if (tile >= numMT * numNT) return;
    const int mt = tile / numNT;
    const int nt = tile - mt * numNT;

    int row = mt * 16 + (lane & 15);
    if (row >= M) row = M - 1;
    const int kb0 = (lane >> 4) * 8;

    v8f acc = {};
#pragma unroll
    for (int kt = 0; kt < KDIM / 32; ++kt) {
        const float* pa = X + (size_t)row * KDIM + kt * 32 + kb0;
        float4 a0 = *(const float4*)(pa);
        float4 a1 = *(const float4*)(pa + 4);
        float4 a2 = *(const float4*)(pa + 16);
        float4 a3 = *(const float4*)(pa + 20);
        v16h a;
        a[0] = (_Float16)a0.x;  a[1] = (_Float16)a0.y;  a[2]  = (_Float16)a0.z;  a[3]  = (_Float16)a0.w;
        a[4] = (_Float16)a1.x;  a[5] = (_Float16)a1.y;  a[6]  = (_Float16)a1.z;  a[7]  = (_Float16)a1.w;
        a[8] = (_Float16)a2.x;  a[9] = (_Float16)a2.y;  a[10] = (_Float16)a2.z;  a[11] = (_Float16)a2.w;
        a[12] = (_Float16)a3.x; a[13] = (_Float16)a3.y; a[14] = (_Float16)a3.z;  a[15] = (_Float16)a3.w;
        v16h b = *(const v16h*)(Bfrag + (size_t)(kt * numNT + nt) * 512 + lane * 16);
        acc = __builtin_amdgcn_wmma_f32_16x16x32_f16(false, a, false, b, (short)0, acc, false, false);
    }
    const int n = nt * 16 + (lane & 15);
    const int m0 = mt * 16 + (lane >> 4) * 8;
    const float bb = bias[n];
#pragma unroll
    for (int r = 0; r < 8; ++r) {
        int m = m0 + r;
        if (m < M) Y[(size_t)m * NDIM + n] = acc[r] + bb;
    }
}

// ---------------------------------------------------------------------------
// 4x4 matmul helper (registers)
// ---------------------------------------------------------------------------
__device__ __forceinline__ void mm4(float* __restrict__ P, const float* __restrict__ A,
                                    const float* __restrict__ B) {
#pragma unroll
    for (int i = 0; i < 4; ++i)
#pragma unroll
        for (int j = 0; j < 4; ++j) {
            float s = 0.f;
#pragma unroll
            for (int k = 0; k < 4; ++k) s = fmaf(A[i * 4 + k], B[k * 4 + j], s);
            P[i * 4 + j] = s;
        }
}

// ---------------------------------------------------------------------------
// Edge kernel: one wave per 16-edge tile. Per-wave LDS (bytes):
//   [0, 8704)    Ae   : [16][272] f16   (concat(h[row],h[col]) staged, 16B-aligned rows)
//   reuse after GEMM1:
//   [0, 2560)    hH   : [16][80]  f16   (relu MLP1 output)
//   [4352, 6400) maps : [16][32]  f32
//   [6400, 8448) Fm   : [2][16][16] f32 (expm results)
//   [8704, 8832) sidx : [32] int        (row idx 0..15, col idx 16..31)
// ---------------------------------------------------------------------------
#define WAVE_SMEM 8960
__global__ void __launch_bounds__(128) sheaf_edge_kernel(
    const float* __restrict__ H, const int* __restrict__ rowIdx, const int* __restrict__ colIdx,
    const _Float16* __restrict__ W1f, const float* __restrict__ b1,
    const _Float16* __restrict__ W2f, const float* __restrict__ b2,
    float* __restrict__ agg, int E_) {
    __shared__ __align__(16) char smem[4][WAVE_SMEM];
    const int lane = threadIdx.x & 31;
    const int wave = threadIdx.x >> 5;
    char* sm = smem[wave];
    _Float16* Ae = (_Float16*)sm;
    int* sidx = (int*)(sm + 8704);

    const int tile = blockIdx.x * 4 + wave;
    if (tile * 16 >= E_) return;
    const int e0 = tile * 16;

    // --- load edge indices (lanes 0..15 row, 16..31 col), prefetch rows ---
    {
        int e = e0 + (lane & 15);
        if (e >= E_) e = E_ - 1;
        int v = (lane < 16) ? rowIdx[e] : colIdx[e];
        sidx[lane] = v;
        __builtin_prefetch(H + (size_t)v * HID, 0, 3);  // global_prefetch_b8
    }

    // --- gather concat(h[row], h[col]) -> f16 LDS tile [16][272] ---
#pragma unroll 4
    for (int i = 0; i < 16; ++i) {
        int r = sidx[i];
        int c = sidx[16 + i];
        const float4* pr = (const float4*)(H + (size_t)r * HID);
        const float4* pc = (const float4*)(H + (size_t)c * HID);
#pragma unroll
        for (int t = 0; t < 2; ++t) {
            int q = t * 32 + lane;                       // float4 index 0..63
            float4 v = (q < 32) ? pr[q] : pc[q - 32];
            _Float16* dst = Ae + i * 272 + q * 4;
            dst[0] = (_Float16)v.x; dst[1] = (_Float16)v.y;
            dst[2] = (_Float16)v.z; dst[3] = (_Float16)v.w;
        }
    }

    const int arow = lane & 15;
    const int kb0 = (lane >> 4) * 8;
    const int m0 = (lane >> 4) * 8;

    // --- GEMM1: Hmlp[16x64] = relu(A[16x256] @ W1 + b1), 32 WMMAs ---
    v8f z = {};
    v8f accH[4] = {z, z, z, z};
#pragma unroll
    for (int kt = 0; kt < 8; ++kt) {
        const _Float16* pa = Ae + arow * 272 + kt * 32 + kb0;
        v8h lo = *(const v8h*)(pa);
        v8h hi = *(const v8h*)(pa + 16);
        v16h a;
#pragma unroll
        for (int t = 0; t < 8; ++t) { a[t] = lo[t]; a[8 + t] = hi[t]; }
#pragma unroll
        for (int nt = 0; nt < 4; ++nt) {
            v16h b = *(const v16h*)(W1f + (size_t)(kt * 4 + nt) * 512 + lane * 16);
            accH[nt] = __builtin_amdgcn_wmma_f32_16x16x32_f16(false, a, false, b, (short)0, accH[nt], false, false);
        }
    }
    // bias + relu, store to LDS as f16 A-operand for GEMM2
    _Float16* hH = (_Float16*)sm;  // [16][80]
#pragma unroll
    for (int nt = 0; nt < 4; ++nt) {
        float bb = b1[nt * 16 + arow];
#pragma unroll
        for (int r = 0; r < 8; ++r) {
            float v = accH[nt][r] + bb;
            v = v > 0.f ? v : 0.f;
            hH[(m0 + r) * 80 + nt * 16 + arow] = (_Float16)v;
        }
    }

    // --- GEMM2: maps[16x32] = Hmlp @ W2 + b2, 4 WMMAs ---
    v8f accM[2] = {z, z};
#pragma unroll
    for (int kt = 0; kt < 2; ++kt) {
        const _Float16* pa = hH + arow * 80 + kt * 32 + kb0;
        v8h lo = *(const v8h*)(pa);
        v8h hi = *(const v8h*)(pa + 16);
        v16h a;
#pragma unroll
        for (int t = 0; t < 8; ++t) { a[t] = lo[t]; a[8 + t] = hi[t]; }
#pragma unroll
        for (int nt = 0; nt < 2; ++nt) {
            v16h b = *(const v16h*)(W2f + (size_t)(kt * 2 + nt) * 512 + lane * 16);
            accM[nt] = __builtin_amdgcn_wmma_f32_16x16x32_f16(false, a, false, b, (short)0, accM[nt], false, false);
        }
    }
    float* mapsL = (float*)(sm + 4352);  // [16][32]
#pragma unroll
    for (int nt = 0; nt < 2; ++nt) {
        float bb = b2[nt * 16 + arow];
#pragma unroll
        for (int r = 0; r < 8; ++r)
            mapsL[(m0 + r) * 32 + nt * 16 + arow] = accM[nt][r] + bb;
    }

    // --- expm of 4x4 skew-symmetric: one matrix per lane (16 edges x 2 maps) ---
    const int eL = lane & 15;   // edge within tile
    const int wL = lane >> 4;   // 0 = f_u, 1 = f_v
    float Aa[16];
    {
        const float* mp = mapsL + eL * 32 + wL * 16;
        float Mm[16];
#pragma unroll
        for (int t = 0; t < 16; ++t) Mm[t] = mp[t];
#pragma unroll
        for (int a = 0; a < 4; ++a)
#pragma unroll
            for (int b = 0; b < 4; ++b) Aa[a * 4 + b] = Mm[a * 4 + b] - Mm[b * 4 + a];
    }
    float T[16];
    {
        float nrm = 0.f;
#pragma unroll
        for (int i = 0; i < 4; ++i) {
            float rs = fabsf(Aa[i * 4]) + fabsf(Aa[i * 4 + 1]) + fabsf(Aa[i * 4 + 2]) + fabsf(Aa[i * 4 + 3]);
            nrm = fmaxf(nrm, rs);
        }
        int j = 0;
        while (nrm > 0.25f && j < 30) { nrm *= 0.5f; ++j; }
        float sc = ldexpf(1.f, -j);
#pragma unroll
        for (int t = 0; t < 16; ++t) Aa[t] *= sc;
        // Horner Taylor, 8 terms: T = I + A(I + A/2(I + ... A/8))
#pragma unroll
        for (int t = 0; t < 16; ++t) T[t] = (t == 0 || t == 5 || t == 10 || t == 15) ? 1.f : 0.f;
#pragma unroll
        for (int k = 8; k >= 1; --k) {
            float P[16];
            mm4(P, Aa, T);
            float inv = 1.f / (float)k;
#pragma unroll
            for (int t = 0; t < 16; ++t)
                T[t] = ((t == 0 || t == 5 || t == 10 || t == 15) ? 1.f : 0.f) + P[t] * inv;
        }
        for (int q = 0; q < j; ++q) {
            float P[16];
            mm4(P, T, T);
#pragma unroll
            for (int t = 0; t < 16; ++t) T[t] = P[t];
        }
    }
    float* Fm = (float*)(sm + 6400);  // [2][16][16]
#pragma unroll
    for (int t = 0; t < 16; ++t) Fm[(wL * 16 + eL) * 16 + t] = T[t];

    // --- projections + scatter: msg = fu^T (fu x_i - fv x_j), agg[col] += msg ---
    float fu[16], fv[16];
#pragma unroll
    for (int t = 0; t < 16; ++t) { fu[t] = Fm[eL * 16 + t]; fv[t] = Fm[(16 + eL) * 16 + t]; }
    const int r_e = sidx[eL];
    const int c_e = sidx[16 + eL];
    const float* xj = H + (size_t)r_e * HID;   // x_row
    const float* xi = H + (size_t)c_e * HID;   // x_col
    float* aggp = agg + (size_t)c_e * HID;
#pragma unroll 2
    for (int kk = 0; kk < 16; ++kk) {
        int k = kk * 2 + wL;                   // k-block 0..31
        float4 vi4 = *(const float4*)(xi + k * DD);
        float4 vj4 = *(const float4*)(xj + k * DD);
        float xiv[4] = {vi4.x, vi4.y, vi4.z, vi4.w};
        float xjv[4] = {vj4.x, vj4.y, vj4.z, vj4.w};
        float err[4];
#pragma unroll
        for (int a = 0; a < 4; ++a) {
            float pi = 0.f, pj = 0.f;
#pragma unroll
            for (int b = 0; b < 4; ++b) {
                pi = fmaf(fu[a * 4 + b], xiv[b], pi);
                pj = fmaf(fv[a * 4 + b], xjv[b], pj);
            }
            err[a] = pi - pj;
        }
#pragma unroll
        for (int a = 0; a < 4; ++a) {
            float m = fmaf(fu[a], err[0],
                      fmaf(fu[4 + a], err[1],
                      fmaf(fu[8 + a], err[2], fu[12 + a] * err[3])));
            atomicAdd(aggp + k * DD + a, m);   // global_atomic_add_f32
        }
    }
}

// ---------------------------------------------------------------------------
// Combine: h_out = elu(h_in - eps * agg)
// ---------------------------------------------------------------------------
__global__ void combine_elu(const float* __restrict__ Hin, const float* __restrict__ Agg,
                            const float* __restrict__ epsP, float* __restrict__ Hout, int total) {
    int i = blockIdx.x * blockDim.x + threadIdx.x;
    if (i >= total) return;
    float eps = epsP[0];
    float v = Hin[i] - eps * Agg[i];
    Hout[i] = v > 0.f ? v : (__expf(v) - 1.f);
}

// ---------------------------------------------------------------------------
extern "C" void kernel_launch(void* const* d_in, const int* in_sizes, int n_in,
                              void* d_out, int out_size, void* d_ws, size_t ws_size,
                              hipStream_t stream) {
    const float* x        = (const float*)d_in[0];
    const int*   ei       = (const int*)d_in[1];
    const float* lin_in_w = (const float*)d_in[2];
    const float* lin_in_b = (const float*)d_in[3];
    const float* c0_w1    = (const float*)d_in[4];
    const float* c0_b1    = (const float*)d_in[5];
    const float* c0_w2    = (const float*)d_in[6];
    const float* c0_b2    = (const float*)d_in[7];
    const float* c0_eps   = (const float*)d_in[8];
    const float* c1_w1    = (const float*)d_in[9];
    const float* c1_b1    = (const float*)d_in[10];
    const float* c1_w2    = (const float*)d_in[11];
    const float* c1_b2    = (const float*)d_in[12];
    const float* c1_eps   = (const float*)d_in[13];
    const float* lo_w     = (const float*)d_in[14];
    const float* lo_b     = (const float*)d_in[15];

    const int N = in_sizes[0] / HID;     // 20000
    const int E = in_sizes[1] / 2;       // 320000
    const int* rowI = ei;
    const int* colI = ei + E;

    // workspace layout: f16 weight fragments first (32B aligned), then f32 buffers
    char* ws = (char*)d_ws;
    _Float16* fLinIn  = (_Float16*)ws;             // 128*128 = 16384 halfs
    _Float16* fW1c0   = fLinIn + 16384;            // 256*64  = 16384
    _Float16* fW2c0   = fW1c0 + 16384;             // 64*32   = 2048
    _Float16* fW1c1   = fW2c0 + 2048;              // 16384
    _Float16* fW2c1   = fW1c1 + 16384;             // 2048
    _Float16* fLinOut = fW2c1 + 2048;              // 128*64  = 8192
    float* h0  = (float*)(ws + 131072);
    float* h1  = h0 + (size_t)N * HID;
    float* agg = h1 + (size_t)N * HID;
    const size_t aggBytes = (size_t)N * HID * sizeof(float);

    // --- pack all weights to WMMA B-fragments ---
    pack_b_f16<<<(128 * 128 + 255) / 256, 256, 0, stream>>>(lin_in_w, fLinIn, 128, 128);
    pack_b_f16<<<(256 * 64 + 255) / 256, 256, 0, stream>>>(c0_w1, fW1c0, 256, 64);
    pack_b_f16<<<(64 * 32 + 255) / 256, 256, 0, stream>>>(c0_w2, fW2c0, 64, 32);
    pack_b_f16<<<(256 * 64 + 255) / 256, 256, 0, stream>>>(c1_w1, fW1c1, 256, 64);
    pack_b_f16<<<(64 * 32 + 255) / 256, 256, 0, stream>>>(c1_w2, fW2c1, 64, 32);
    pack_b_f16<<<(128 * 64 + 255) / 256, 256, 0, stream>>>(lo_w, fLinOut, 128, 64);

    // --- lin_in: h0 = x @ W + b ---
    {
        int tiles = ((N + 15) / 16) * (128 / 16);
        node_gemm<128, 128><<<(tiles + 7) / 8, 256, 0, stream>>>(x, fLinIn, lin_in_b, h0, N);
    }

    const int edgeTiles = (E + 15) / 16;
    const int edgeBlocks = (edgeTiles + 3) / 4;
    const int combBlocks = ((int)((size_t)N * HID) + 255) / 256;

    // --- conv0 ---
    hipMemsetAsync(agg, 0, aggBytes, stream);
    sheaf_edge_kernel<<<edgeBlocks, 128, 0, stream>>>(h0, rowI, colI, fW1c0, c0_b1, fW2c0, c0_b2, agg, E);
    combine_elu<<<combBlocks, 256, 0, stream>>>(h0, agg, c0_eps, h1, N * HID);

    // --- conv1 ---
    hipMemsetAsync(agg, 0, aggBytes, stream);
    sheaf_edge_kernel<<<edgeBlocks, 128, 0, stream>>>(h1, rowI, colI, fW1c1, c1_b1, fW2c1, c1_b2, agg, E);
    combine_elu<<<combBlocks, 256, 0, stream>>>(h1, agg, c1_eps, h0, N * HID);

    // --- lin_out: out = h @ W + b ---
    {
        int tiles = ((N + 15) / 16) * (64 / 16);
        node_gemm<128, 64><<<(tiles + 7) / 8, 256, 0, stream>>>(h0, fLinOut, lo_b, (float*)d_out, N);
    }
}
